// SpatiallyMaskedSelfAttention_Light_89464168775865
// MI455X (gfx1250) — matrix operand order repaired and checked by reading, not compiled, not run on hardware
//
#include <hip/hip_runtime.h>
#include <hip/hip_bf16.h>

// ---------------------------------------------------------------------------
// SpatiallyMaskedSelfAttention for MI455X (gfx1250, wave32, WMMA)
//   x:[8,1024,768] f32, qkv_w:[2304,768] f32, proj_w:[768,768] f32
//   mask: d2((yi,xi),(yj,xj)) > 16 -> -1e9   (computed analytically)
// Pipeline: f32->f16 cast -> QKV GEMM (32x64 tile/wave, WMMA) -> flash
// attention (WMMA, online softmax, LDS relayout of P) -> proj GEMM -> f32 out
// ---------------------------------------------------------------------------

typedef __attribute__((ext_vector_type(16))) _Float16 v16h;
typedef __attribute__((ext_vector_type(8)))  float    v8f;

#define DIMC 768
#define NHEADS 12
#define HDIM 64
#define SEQ 1024
#define BATCH 8
#define QKVO 2304
#define MROWS (BATCH * SEQ)        // 8192

__device__ __forceinline__ v8f wmma_f16(v16h a, v16h b, v8f c) {
  return __builtin_amdgcn_wmma_f32_16x16x32_f16(false, a, false, b, (short)0, c,
                                                false, false);
}

// A-fragment (16x32 f16). 'row' already points at src + m*ld + kb where
// m = lane&15, kb = (lane<16)?0:16. Lane holds K = kb+{0..15}.
__device__ __forceinline__ v16h load_frag_a(const _Float16* __restrict__ row,
                                            int k0) {
  v16h a;
#pragma unroll
  for (int j = 0; j < 8; ++j) {
    a[j]     = row[k0 + j];
    a[j + 8] = row[k0 + 8 + j];
  }
  return a;
}

// B-fragment (32x16 f16). 'p' points at 16 contiguous K values for this
// lane's output column n = lane&15, starting at K = (lane<16)?0:16.
__device__ __forceinline__ v16h load_frag_b(const _Float16* __restrict__ p) {
  v16h b;
#pragma unroll
  for (int j = 0; j < 16; ++j) b[j] = p[j];
  return b;
}

// 32x64 output tile per wave: 2 A-fragments x 4 B-fragments -> 8 WMMA/k-step.
// acc[t] = rows mt*32+0..15, cols nt*64+t*16..; acc[4+t] = rows +16..31.
__device__ __forceinline__ void gemm_tile_32x64(
    const _Float16* __restrict__ A, const _Float16* __restrict__ W, int mt,
    int nt, int lane, v8f acc[8]) {
  const int laneN = lane & 15;
  const int kbA = (lane < 16) ? 0 : 16;
  const int kbB = (lane < 16) ? 0 : 16;

  const _Float16* ar0 = A + (size_t)(mt * 32 + laneN) * DIMC + kbA;
  const _Float16* ar1 = ar0 + (size_t)16 * DIMC;
  const _Float16* br0 = W + (size_t)(nt * 64 + laneN) * DIMC + kbB;
  const _Float16* br1 = br0 + (size_t)16 * DIMC;
  const _Float16* br2 = br0 + (size_t)32 * DIMC;
  const _Float16* br3 = br0 + (size_t)48 * DIMC;

  for (int k0 = 0; k0 < DIMC; k0 += 32) {
    __builtin_prefetch(ar0 + k0 + 128, 0, 1);
    __builtin_prefetch(ar1 + k0 + 128, 0, 1);
    const v16h a0 = load_frag_a(ar0, k0);
    const v16h a1 = load_frag_a(ar1, k0);
    {
      const v16h b = load_frag_b(br0 + k0);
      acc[0] = wmma_f16(a0, b, acc[0]);
      acc[4] = wmma_f16(a1, b, acc[4]);
    }
    {
      const v16h b = load_frag_b(br1 + k0);
      acc[1] = wmma_f16(a0, b, acc[1]);
      acc[5] = wmma_f16(a1, b, acc[5]);
    }
    {
      const v16h b = load_frag_b(br2 + k0);
      acc[2] = wmma_f16(a0, b, acc[2]);
      acc[6] = wmma_f16(a1, b, acc[6]);
    }
    {
      const v16h b = load_frag_b(br3 + k0);
      acc[3] = wmma_f16(a0, b, acc[3]);
      acc[7] = wmma_f16(a1, b, acc[7]);
    }
  }
}

// ------------------------------- f32 -> f16 --------------------------------
__global__ void cvt_f32_f16(const float* __restrict__ s,
                            _Float16* __restrict__ d, int n) {
  int i = blockIdx.x * blockDim.x + threadIdx.x;
  if (i < n) d[i] = (_Float16)s[i];
}

// ------------------------------- QKV GEMM ----------------------------------
// C[m,o] = sum_c xh[m,c] * wqkv[o,c]; scatter into q/k/vT (f16).
// 64-wide N-tiles align with q/k/v and head boundaries -> uniform epilogue.
__global__ __launch_bounds__(128) void qkv_gemm(
    const _Float16* __restrict__ xh, const _Float16* __restrict__ wh,
    _Float16* __restrict__ q, _Float16* __restrict__ kk,
    _Float16* __restrict__ vT) {
  const int lane  = threadIdx.x & 31;
  const int wave  = threadIdx.x >> 5;
  const int tile  = blockIdx.x * 4 + wave;
  const int ntile = QKVO / 64;                  // 36
  const int mt = tile / ntile, nt = tile % ntile;
  const int laneN   = lane & 15;
  const int rowBase = (lane < 16) ? 0 : 8;

  v8f acc[8];
#pragma unroll
  for (int i = 0; i < 8; ++i) acc[i] = {};
  gemm_tile_32x64(xh, wh, mt, nt, lane, acc);

  const int n0 = nt * 64;
  const int which = n0 / DIMC;                  // uniform per tile
  const int h = (n0 % DIMC) / HDIM;             // uniform per tile
#pragma unroll
  for (int sub = 0; sub < 2; ++sub) {
#pragma unroll
    for (int t = 0; t < 4; ++t) {
      const int d = t * 16 + laneN;
#pragma unroll
      for (int e = 0; e < 8; ++e) {
        const int m  = mt * 32 + sub * 16 + rowBase + e;
        const int b_ = m >> 10, ns = m & (SEQ - 1);
        const _Float16 val = (_Float16)acc[sub * 4 + t][e];
        if (which == 0)
          q[(((size_t)(b_ * NHEADS + h) * SEQ) + ns) * HDIM + d] = val;
        else if (which == 1)
          kk[(((size_t)(b_ * NHEADS + h) * SEQ) + ns) * HDIM + d] = val;
        else
          vT[(((size_t)(b_ * NHEADS + h) * HDIM) + d) * SEQ + ns] = val;
      }
    }
  }
}

// ---------------------------- Flash attention ------------------------------
// grid: (SEQ/64, B*H). Each wave owns 16 query rows, streams 32 key columns
// per step. Score tiles via WMMA, analytic spatial mask, online softmax,
// P relayout C-layout -> A-layout through per-wave LDS, P*V via WMMA on vT.
__global__ __launch_bounds__(128) void flash_attn(
    const _Float16* __restrict__ q, const _Float16* __restrict__ k,
    const _Float16* __restrict__ vT, _Float16* __restrict__ attn_out) {
  __shared__ _Float16 pbuf[4][16 * 32];

  const int lane = threadIdx.x & 31;
  const int wave = threadIdx.x >> 5;
  const int bh = blockIdx.y;
  const int b = bh / NHEADS, h = bh % NHEADS;
  const int m0 = blockIdx.x * 64 + wave * 16;

  const int laneN   = lane & 15;
  const int rowBase = (lane < 16) ? 0 : 8;
  const int kbA     = (lane < 16) ? 0 : 16;
  const int kbB     = (lane < 16) ? 0 : 16;

  const _Float16* qbase = q + (size_t)bh * SEQ * HDIM;
  const _Float16* kbase = k + (size_t)bh * SEQ * HDIM;
  const _Float16* vbase = vT + (size_t)bh * HDIM * SEQ;

  // Q fragments (16 rows x K=64) stay in registers.
  const _Float16* qrow = qbase + (size_t)(m0 + laneN) * HDIM + kbA;
  const v16h qa0 = load_frag_a(qrow, 0);
  const v16h qa1 = load_frag_a(qrow, 32);

  int yi[8], xi[8];
#pragma unroll
  for (int e = 0; e < 8; ++e) {
    const int i = m0 + rowBase + e;
    yi[e] = i >> 5;
    xi[e] = i & 31;
  }

  float m_r[8], l_r[8];
  v8f oacc[4];
#pragma unroll
  for (int e = 0; e < 8; ++e) { m_r[e] = -1e30f; l_r[e] = 0.f; }
#pragma unroll
  for (int t = 0; t < 4; ++t) oacc[t] = {};

  const float scale = 0.125f;  // 64^-0.5

  for (int c0 = 0; c0 < SEQ; c0 += 32) {
    if (c0 + 32 < SEQ) {
      __builtin_prefetch(kbase + (size_t)(c0 + 32 + laneN) * HDIM, 0, 1);
      __builtin_prefetch(kbase + (size_t)(c0 + 48 + laneN) * HDIM, 0, 1);
    }
    // ---- scores: S[16x32] = Q(16x64) * K(cols c0..c0+31)^T ----
    const _Float16* kr0 = kbase + (size_t)(c0 + laneN) * HDIM + kbB;
    const _Float16* kr1 = kbase + (size_t)(c0 + 16 + laneN) * HDIM + kbB;
    v8f s0 = {}, s1 = {};
    s0 = wmma_f16(qa0, load_frag_b(kr0), s0);
    s0 = wmma_f16(qa1, load_frag_b(kr0 + 32), s0);
    s1 = wmma_f16(qa0, load_frag_b(kr1), s1);
    s1 = wmma_f16(qa1, load_frag_b(kr1 + 32), s1);

    // ---- scale + spatial mask (d2 > 16 -> -1e9) ----
    const int j0 = c0 + laneN, j1 = j0 + 16;
    const int yj0 = j0 >> 5, xj0 = j0 & 31;
    const int yj1 = j1 >> 5, xj1 = j1 & 31;
#pragma unroll
    for (int e = 0; e < 8; ++e) {
      float a = s0[e] * scale, c = s1[e] * scale;
      const int dy0 = yi[e] - yj0, dx0 = xi[e] - xj0;
      const int dy1 = yi[e] - yj1, dx1 = xi[e] - xj1;
      if (dy0 * dy0 + dx0 * dx0 > 16) a = -1e9f;
      if (dy1 * dy1 + dx1 * dx1 > 16) c = -1e9f;
      s0[e] = a;
      s1[e] = c;
    }

    // ---- online softmax (rows live across each 16-lane group) ----
    float p0[8], p1[8];
#pragma unroll
    for (int e = 0; e < 8; ++e) {
      float mx = fmaxf(s0[e], s1[e]);
#pragma unroll
      for (int off = 1; off < 16; off <<= 1)
        mx = fmaxf(mx, __shfl_xor(mx, off, 32));
      const float mnew = fmaxf(m_r[e], mx);
      const float alpha = __expf(m_r[e] - mnew);
      p0[e] = __expf(s0[e] - mnew);
      p1[e] = __expf(s1[e] - mnew);
      float rs = p0[e] + p1[e];
#pragma unroll
      for (int off = 1; off < 16; off <<= 1) rs += __shfl_xor(rs, off, 32);
      l_r[e] = l_r[e] * alpha + rs;
      m_r[e] = mnew;
#pragma unroll
      for (int t = 0; t < 4; ++t) oacc[t][e] *= alpha;
    }

    // ---- P: C-layout -> A-layout via per-wave LDS tile (16x32 f16) ----
#pragma unroll
    for (int e = 0; e < 8; ++e) {
      pbuf[wave][(rowBase + e) * 32 + laneN]      = (_Float16)p0[e];
      pbuf[wave][(rowBase + e) * 32 + 16 + laneN] = (_Float16)p1[e];
    }
    asm volatile("s_wait_dscnt 0" ::: "memory");
    v16h pa;
#pragma unroll
    for (int j = 0; j < 8; ++j) {
      pa[j]     = pbuf[wave][laneN * 32 + kbA + j];
      pa[j + 8] = pbuf[wave][laneN * 32 + 8 + kbA + j];
    }

    // ---- O += P(16x32) * V(rows c0..c0+31) using vT rows ----
#pragma unroll
    for (int t = 0; t < 4; ++t) {
      const _Float16* vr = vbase + (size_t)(t * 16 + laneN) * SEQ + c0 + kbB;
      oacc[t] = wmma_f16(pa, load_frag_b(vr), oacc[t]);
    }
  }

  float inv[8];
#pragma unroll
  for (int e = 0; e < 8; ++e) inv[e] = 1.0f / l_r[e];

#pragma unroll
  for (int t = 0; t < 4; ++t) {
#pragma unroll
    for (int e = 0; e < 8; ++e) {
      const int i = m0 + rowBase + e;
      attn_out[((size_t)(b * SEQ + i)) * DIMC + h * HDIM + t * 16 + laneN] =
          (_Float16)(oacc[t][e] * inv[e]);
    }
  }
}

// ----------------------------- Projection GEMM -----------------------------
__global__ __launch_bounds__(128) void proj_gemm(
    const _Float16* __restrict__ ah, const _Float16* __restrict__ wh,
    float* __restrict__ out) {
  const int lane  = threadIdx.x & 31;
  const int wave  = threadIdx.x >> 5;
  const int tile  = blockIdx.x * 4 + wave;
  const int ntile = DIMC / 64;                  // 12
  const int mt = tile / ntile, nt = tile % ntile;
  const int laneN   = lane & 15;
  const int rowBase = (lane < 16) ? 0 : 8;

  v8f acc[8];
#pragma unroll
  for (int i = 0; i < 8; ++i) acc[i] = {};
  gemm_tile_32x64(ah, wh, mt, nt, lane, acc);

#pragma unroll
  for (int sub = 0; sub < 2; ++sub) {
#pragma unroll
    for (int t = 0; t < 4; ++t) {
      const int o = nt * 64 + t * 16 + laneN;
#pragma unroll
      for (int e = 0; e < 8; ++e) {
        const int m = mt * 32 + sub * 16 + rowBase + e;
        out[(size_t)m * DIMC + o] = acc[sub * 4 + t][e];
      }
    }
  }
}

// ------------------------------- launcher ----------------------------------
extern "C" void kernel_launch(void* const* d_in, const int* in_sizes, int n_in,
                              void* d_out, int out_size, void* d_ws,
                              size_t ws_size, hipStream_t stream) {
  const float* x      = (const float*)d_in[0];  // [8,1024,768]
  const float* qkv_w  = (const float*)d_in[1];  // [2304,768]
  const float* proj_w = (const float*)d_in[2];  // [768,768]
  // d_in[3] spatial_mask unused (computed analytically)

  const size_t n_x    = (size_t)MROWS * DIMC;   // 6291456
  const size_t n_wqkv = (size_t)QKVO * DIMC;    // 1769472
  const size_t n_wprj = (size_t)DIMC * DIMC;    // 589824
  const size_t n_head = (size_t)BATCH * NHEADS * SEQ * HDIM;  // 6291456

  _Float16* p = (_Float16*)d_ws;
  _Float16* xh    = p; p += n_x;
  _Float16* wqkvh = p; p += n_wqkv;
  _Float16* wprjh = p; p += n_wprj;
  _Float16* qb    = p; p += n_head;
  _Float16* kb    = p; p += n_head;
  _Float16* vTb   = p; p += n_head;
  _Float16* attnb = p; p += n_x;

  cvt_f32_f16<<<(int)((n_x + 255) / 256), 256, 0, stream>>>(x, xh, (int)n_x);
  cvt_f32_f16<<<(int)((n_wqkv + 255) / 256), 256, 0, stream>>>(qkv_w, wqkvh,
                                                               (int)n_wqkv);
  cvt_f32_f16<<<(int)((n_wprj + 255) / 256), 256, 0, stream>>>(proj_w, wprjh,
                                                               (int)n_wprj);

  // (8192/32)*(2304/64) = 9216 tiles, 4 waves/block
  qkv_gemm<<<9216 / 4, 128, 0, stream>>>(xh, wqkvh, qb, kb, vTb);

  flash_attn<<<dim3(SEQ / 64, BATCH * NHEADS), 128, 0, stream>>>(qb, kb, vTb,
                                                                 attnb);

  // (8192/32)*(768/64) = 3072 tiles
  proj_gemm<<<3072 / 4, 128, 0, stream>>>(attnb, wprjh, (float*)d_out);
}
